// FC_quant_86234353369428
// MI455X (gfx1250) — matrix operand order repaired and checked by reading, not compiled, not run on hardware
//
#include <hip/hip_runtime.h>
#include <stdint.h>

// ---------------------------------------------------------------------------
// Binarized FC: Z = X @ (sign(W)*WERR) + sign(bias)*BERR
//   X[16384,4096] f32, W[4096,1024] f32, bias[1,1024] f32 -> Z[16384,1024] f32
// bf16 WMMA GEMM, double-buffered LDS, async global->LDS for the B tile,
// native bf16 converts for the A tile, NT stores for Z.
// ---------------------------------------------------------------------------

#define IN_F   4096
#define OUT_F  1024
#define BATCH  16384
#define BERR_C 1.0f

#define BM   128
#define BN   128
#define BK   64          // two 16x16x32 WMMA K-substeps per stage
#define NK   (IN_F / BK) // 64 stages
#define LDSS 72          // u16 per LDS row: 64 data + 8 pad (144B, 16B-aligned)

typedef __bf16          v16bf  __attribute__((ext_vector_type(16)));
typedef __bf16          bf16x8 __attribute__((ext_vector_type(8)));
typedef __bf16          bf16x4 __attribute__((ext_vector_type(4)));
typedef float           v8f    __attribute__((ext_vector_type(8)));
typedef unsigned short  u16x16 __attribute__((ext_vector_type(16)));
typedef unsigned short  u16x8  __attribute__((ext_vector_type(8)));
typedef float           f32x4  __attribute__((ext_vector_type(4)));
typedef int             v4i_vs __attribute__((vector_size(16)));   // matches builtin param

#define AS1 __attribute__((address_space(1)))
#define AS3 __attribute__((address_space(3)))

// ---------------------------------------------------------------------------
// Kernel 1: binarize W and transpose into WbT[N][K] (bf16 stored as u16).
// ---------------------------------------------------------------------------
__global__ __launch_bounds__(256)
void binarize_transpose_w(const float* __restrict__ W,
                          unsigned short* __restrict__ WbT) {
    int idx = blockIdx.x * 256 + threadIdx.x;       // over IN_F*OUT_F
    int k = idx >> 10;                              // / OUT_F
    int n = idx & (OUT_F - 1);
    float w = W[idx];
    unsigned short s = (w > 0.0f) ? (unsigned short)0x3F80
                     : (w < 0.0f) ? (unsigned short)0xBF80
                                  : (unsigned short)0x0000;
    WbT[(size_t)n * IN_F + k] = s;
}

// ---------------------------------------------------------------------------
// Staging helpers
// ---------------------------------------------------------------------------
// B tile: 4x async 16B transfers per thread, no VGPR round-trip (ASYNCcnt).
__device__ __forceinline__ void issue_B_async(const unsigned short* srcB,
                                              unsigned short* ldsB) {
    #pragma unroll
    for (int c = 0; c < 4; ++c)
        __builtin_amdgcn_global_load_async_to_lds_b128(
            (AS1 v4i_vs*)(srcB + c * 8),
            (AS3 v4i_vs*)(ldsB + c * 8), 0, 0);
}

// A tile: load 32 f32 into regs (issue early, commit later).
__device__ __forceinline__ void load_A_regs(const float* srcA, f32x4 areg[8]) {
    #pragma unroll
    for (int c = 0; c < 8; ++c) areg[c] = *(const f32x4*)(srcA + c * 4);
}

// Native f32->bf16 converts (v_cvt_pk_bf16_f32 class) + 4x ds_store_b128.
__device__ __forceinline__ void commit_A(const f32x4 areg[8],
                                         unsigned short* ldsA) {
    #pragma unroll
    for (int c = 0; c < 4; ++c) {
        bf16x4 lo = __builtin_convertvector(areg[2 * c + 0], bf16x4);
        bf16x4 hi = __builtin_convertvector(areg[2 * c + 1], bf16x4);
        bf16x8 p;
        #pragma unroll
        for (int e = 0; e < 4; ++e) { p[e] = lo[e]; p[e + 4] = hi[e]; }
        *(bf16x8*)(ldsA + c * 8) = p;
    }
}

// ---------------------------------------------------------------------------
// Kernel 2: bf16 WMMA GEMM.  Block = 256 threads = 8 waves (wave32).
// Block tile 128x128, K-stage 64.  Wave tile 32x64 = 2x4 WMMA tiles
// -> 16 v_wmma_f32_16x16x32_bf16 per stage, 1 barrier per stage.
// ---------------------------------------------------------------------------
__global__ __launch_bounds__(256)
void bgemm_wmma_bf16(const float* __restrict__ X,
                     const unsigned short* __restrict__ WbT,
                     const float* __restrict__ bias,
                     float* __restrict__ Z) {
    __shared__ __attribute__((aligned(16))) unsigned short AsAll[2 * BM * LDSS];
    __shared__ __attribute__((aligned(16))) unsigned short BsAll[2 * BN * LDSS];

    const int tid  = threadIdx.x;
    const int lane = tid & 31;
    const int wave = tid >> 5;
    const int wm   = wave & 3;          // M group 0..3 (32 rows)
    const int wn   = wave >> 2;         // N group 0..1 (64 cols)

    const int m_blk = blockIdx.y * BM;
    const int n_blk = blockIdx.x * BN;

    // global->LDS staging: 2 threads per row, 32 elements each
    const int g_row  = tid >> 1;        // 0..127
    const int g_half = (tid & 1) * 32;  // k offset 0 / 32

    const int lane_lo = lane & 15;
    const int lane_hi = lane >> 4;      // 0 or 1
    const int ak = lane_hi * 8;         // A-fragment K chunk select
    const int bk = lane_hi * 8;         // B-fragment K chunk select

    v8f acc[2][4];
    #pragma unroll
    for (int i = 0; i < 2; ++i)
        #pragma unroll
        for (int j = 0; j < 4; ++j)
            #pragma unroll
            for (int e = 0; e < 8; ++e) acc[i][j][e] = 0.0f;

    const float*          srcA0 = X   + (size_t)(m_blk + g_row) * IN_F + g_half;
    const unsigned short* srcB0 = WbT + (size_t)(n_blk + g_row) * IN_F + g_half;
    unsigned short* const ldsA  = &AsAll[g_row * LDSS + g_half];
    unsigned short* const ldsB  = &BsAll[g_row * LDSS + g_half];

    // ---- preload stage 0 into buffer 0 ----
    {
        issue_B_async(srcB0, ldsB);
        f32x4 areg[8];
        load_A_regs(srcA0, areg);
        commit_A(areg, ldsA);
        asm volatile("s_wait_asynccnt 0x0" ::: "memory");
        __syncthreads();
    }

    int buf = 0;
    for (int it = 0; it < NK; ++it) {
        const int knext = (it + 1) * BK;
        const int nbuf  = buf ^ 1;

        // ---- issue next-stage loads (async B straight to LDS, A to regs) ----
        f32x4 areg[8];
        if (knext < IN_F) {
            issue_B_async(srcB0 + knext, ldsB + nbuf * BN * LDSS);
            load_A_regs(srcA0 + knext, areg);
        }

        // ---- compute current stage: 2 K-substeps x 8 WMMA ----
        const unsigned short* As = &AsAll[buf * BM * LDSS];
        const unsigned short* Bs = &BsAll[buf * BN * LDSS];
        #pragma unroll
        for (int s = 0; s < 2; ++s) {
            const int kb = s * 32;

            // A frag: lanes 0-15 elems 0..7=K0..7, 8..15=K16..23 (M=lane);
            //         lanes 16-31 elems 0..7=K8..15, 8..15=K24..31
            v16bf afrag[2];
            #pragma unroll
            for (int mt = 0; mt < 2; ++mt) {
                const int r = wm * 32 + mt * 16 + lane_lo;
                u16x8 lo = *(const u16x8*)&As[r * LDSS + kb + ak + 0];
                u16x8 hi = *(const u16x8*)&As[r * LDSS + kb + ak + 16];
                u16x16 a;
                #pragma unroll
                for (int e = 0; e < 8; ++e) { a[e] = lo[e]; a[e + 8] = hi[e]; }
                afrag[mt] = __builtin_bit_cast(v16bf, a);
            }
            // B frag: lane = column N (mod 16); lanes 0-15 K0..7,K16..23; 16-31 K8..15,K24..31
            v16bf bfrag[4];
            #pragma unroll
            for (int nt = 0; nt < 4; ++nt) {
                const int n = wn * 64 + nt * 16 + lane_lo;
                u16x8 lo = *(const u16x8*)&Bs[n * LDSS + kb + bk + 0];
                u16x8 hi = *(const u16x8*)&Bs[n * LDSS + kb + bk + 16];
                u16x16 b;
                #pragma unroll
                for (int e = 0; e < 8; ++e) { b[e] = lo[e]; b[e + 8] = hi[e]; }
                bfrag[nt] = __builtin_bit_cast(v16bf, b);
            }
            #pragma unroll
            for (int mt = 0; mt < 2; ++mt)
                #pragma unroll
                for (int nt = 0; nt < 4; ++nt)
                    acc[mt][nt] = __builtin_amdgcn_wmma_f32_16x16x32_bf16(
                        false, afrag[mt], false, bfrag[nt],
                        (short)0, acc[mt][nt], false, false);
        }

        // ---- commit next A tile (f32->bf16), then sync buffers ----
        if (knext < IN_F)
            commit_A(areg, ldsA + nbuf * BM * LDSS);

        asm volatile("s_wait_asynccnt 0x0" ::: "memory");
        __syncthreads();
        buf = nbuf;
    }

    // ---- epilogue: D layout (VGPR r -> M=r lanes 0-15, M=8+r lanes 16-31) ----
    // Z is write-once -> nontemporal stores, keep L2 for X/WbT reuse.
    #pragma unroll
    for (int nt = 0; nt < 4; ++nt) {
        const int n = n_blk + wn * 64 + nt * 16 + lane_lo;
        const float bv = bias[n];
        const float bb = ((bv > 0.0f) ? 1.0f : (bv < 0.0f) ? -1.0f : 0.0f) * BERR_C;
        #pragma unroll
        for (int mt = 0; mt < 2; ++mt) {
            const int mbase = m_blk + wm * 32 + mt * 16 + lane_hi * 8;
            #pragma unroll
            for (int r = 0; r < 8; ++r)
                __builtin_nontemporal_store(acc[mt][nt][r] + bb,
                                            &Z[(size_t)(mbase + r) * OUT_F + n]);
        }
    }
}

// ---------------------------------------------------------------------------
extern "C" void kernel_launch(void* const* d_in, const int* in_sizes, int n_in,
                              void* d_out, int out_size, void* d_ws, size_t ws_size,
                              hipStream_t stream) {
    const float* X    = (const float*)d_in[0];   // [16384,4096]
    const float* W    = (const float*)d_in[1];   // [4096,1024]
    const float* bias = (const float*)d_in[2];   // [1,1024]
    float*       Z    = (float*)d_out;           // [16384,1024]
    unsigned short* WbT = (unsigned short*)d_ws; // [1024,4096] bf16 = 8 MB

    // 1) binarize + transpose W -> bf16
    {
        const int total  = IN_F * OUT_F;         // 4,194,304
        binarize_transpose_w<<<total / 256, 256, 0, stream>>>(W, WbT);
    }
    // 2) WMMA GEMM
    {
        dim3 grid(OUT_F / BN, BATCH / BM);       // (8, 128)
        bgemm_wmma_bf16<<<grid, 256, 0, stream>>>(X, WbT, bias, Z);
    }
}
